// SAGE_13657996001661
// MI455X (gfx1250) — matrix operand order repaired and checked by reading, not compiled, not run on hardware
//
#include <hip/hip_runtime.h>
#include <hip/hip_bf16.h>

typedef __attribute__((ext_vector_type(16))) __bf16 v16bf;
typedef __attribute__((ext_vector_type(8)))  float  v8f;

// ---------------------------------------------------------------------------
// Small utility kernels
// ---------------------------------------------------------------------------
__global__ void zero_kernel(float* __restrict__ p, long n) {
  long i = (long)blockIdx.x * blockDim.x + threadIdx.x;
  if (i < n) p[i] = 0.f;
}

// Pad x [N,100] -> h [N,128] (cols 100..127 = 0)
__global__ void pad_x_kernel(const float* __restrict__ x, float* __restrict__ h,
                             long total, int N) {
  long i = (long)blockIdx.x * blockDim.x + threadIdx.x;
  if (i >= total) return;
  int node = (int)(i >> 7), d = (int)(i & 127);
  h[i] = (d < 100) ? x[(long)node * 100 + d] : 0.f;
}

// Pack f32 weight [K,O] into bf16 WMMA B-fragment order, zero-padded:
// flat index i = (((kc*tiles + ct)*32 + lane)*16 + e)
//   k = kc*32 + (e>>3)*16 + (lane>>4)*8 + (e&7) ; n = ct*16 + (lane&15)
__global__ void pack_w_kernel(const float* __restrict__ src, int K, int O,
                              __bf16* __restrict__ dst, int tiles) {
  int i = blockIdx.x * blockDim.x + threadIdx.x;
  int total = 4 * tiles * 512;
  if (i >= total) return;
  int e = i & 15;
  int lane = (i >> 4) & 31;
  int rest = i >> 9;
  int ct = rest % tiles;
  int kc = rest / tiles;
  int k = kc * 32 + (e >> 3) * 16 + (lane >> 4) * 8 + (e & 7);
  int n = ct * 16 + (lane & 15);
  float v = (k < K && n < O) ? src[k * O + n] : 0.f;
  dst[i] = (__bf16)v;
}

// deg[dst] += 1 per edge
__global__ void deg_kernel(const int* __restrict__ dst, float* __restrict__ deg, int E) {
  int i = blockIdx.x * blockDim.x + threadIdx.x;
  if (i < E) atomicAdd(&deg[dst[i]], 1.f);
}

// agg[dst] += h[src]; one thread per (edge, 4-float chunk)
__global__ void scatter_kernel(const float* __restrict__ h, const int* __restrict__ src,
                               const int* __restrict__ dst, float* __restrict__ agg,
                               long total) {
  long i = (long)blockIdx.x * blockDim.x + threadIdx.x;
  if (i >= total) return;
  int e = (int)(i >> 5);
  int part = (int)(i & 31);
  int s = src[e], d = dst[e];
  const float* hp = h + (long)s * 128 + part * 4;
  float* ap = agg + (long)d * 128 + part * 4;
  atomicAdd(ap + 0, hp[0]);
  atomicAdd(ap + 1, hp[1]);
  atomicAdd(ap + 2, hp[2]);
  atomicAdd(ap + 3, hp[3]);
}

// agg /= max(deg,1)
__global__ void div_kernel(float* __restrict__ agg, const float* __restrict__ deg, long total) {
  long i = (long)blockIdx.x * blockDim.x + threadIdx.x;
  if (i >= total) return;
  agg[i] = agg[i] / fmaxf(deg[i >> 7], 1.f);
}

// ---------------------------------------------------------------------------
// Fused SAGE GEMM:  out = A @ Bw (+ Ar @ Br) + bias, optional BN(eval)+ReLU
// A, Ar: f32 [nrows,128]; Bw, Br: bf16 pre-packed fragment order (see pack_w).
// Block = 128 threads (4 wave32s) covering 64 rows; wave w owns rows 16w..16w+15
// and all TILES column tiles (TILES v8f accumulators held in VGPRs).
// ---------------------------------------------------------------------------
template <int TILES, bool DUAL>
__global__ void __launch_bounds__(128)
sage_gemm(const float* __restrict__ A, const float* __restrict__ Ar,
          const __bf16* __restrict__ Bw, const __bf16* __restrict__ Br,
          const float* __restrict__ bias,
          const float* __restrict__ bn_g, const float* __restrict__ bn_b,
          const float* __restrict__ bn_m, const float* __restrict__ bn_v,
          int relu, int nrows, float* __restrict__ out, int outCols) {
  // A fragments staged in LDS already in WMMA fragment order (bf16):
  // sA[wavetile][lane][e], 4*32*16 bf16 = 4 KB each
  __shared__ __attribute__((aligned(32))) __bf16 sA[4 * 32 * 16];
  __shared__ __attribute__((aligned(32))) __bf16 sX[4 * 32 * 16];

  const int tid = threadIdx.x;
  const int wave = tid >> 5;
  const int lane = tid & 31;
  const int hi = lane >> 4;   // lane half
  const int nn = lane & 15;   // column within tile / row within tile
  const int rowBase = blockIdx.x * 64;

  v8f acc[TILES];
  v8f zv = {0.f, 0.f, 0.f, 0.f, 0.f, 0.f, 0.f, 0.f};
#pragma unroll
  for (int i = 0; i < TILES; ++i) acc[i] = zv;

#pragma unroll
  for (int kci = 0; kci < 4; ++kci) {
    const int kc = kci * 32;
    // Cooperative coalesced load of the 64x32 f32 chunk; write bf16 into LDS
    // directly in fragment order via the inverse ISA A-layout map.
    for (int i = tid; i < 64 * 32; i += 128) {
      int r = i >> 5, c = i & 31;
      int gr = rowBase + r;
      float va = 0.f, vx = 0.f;
      if (gr < nrows) {
        va = A[(long)gr * 128 + kc + c];
        if (DUAL) vx = Ar[(long)gr * 128 + kc + c];
      }
      // inverse of k = (p&4)*4 + hi*8 + (p&3)*2 + b
      int b = c & 1;
      int fh = (c >> 3) & 1;
      int p = ((c >> 4) << 2) | ((c >> 1) & 3);
      int e = (p << 1) | b;
      int flane = fh * 16 + (r & 15);
      int widx = (((r >> 4) * 32 + flane) << 4) + e;
      sA[widx] = (__bf16)va;
      if (DUAL) sX[widx] = (__bf16)vx;
    }
    __syncthreads();

    const v16bf aA = *(const v16bf*)&sA[(wave * 32 + lane) << 4];
    v16bf aX;
    if (DUAL) aX = *(const v16bf*)&sX[(wave * 32 + lane) << 4];

    const __bf16* bw = Bw + (long)(kci * TILES) * 512 + lane * 16;
    const __bf16* br = DUAL ? (Br + (long)(kci * TILES) * 512 + lane * 16) : nullptr;
#pragma unroll
    for (int ct = 0; ct < TILES; ++ct) {
      const v16bf bW = *(const v16bf*)(bw + ct * 512);
      acc[ct] = __builtin_amdgcn_wmma_f32_16x16x32_bf16(
          false, aA, false, bW, (short)0, acc[ct], false, false);
      if (DUAL) {
        const v16bf bR = *(const v16bf*)(br + ct * 512);
        acc[ct] = __builtin_amdgcn_wmma_f32_16x16x32_bf16(
            false, aX, false, bR, (short)0, acc[ct], false, false);
      }
    }
    __syncthreads();
  }

  // Epilogue: D layout — VGPR r holds row M = (lane>>4)*8 + r, col N = lane&15
#pragma unroll
  for (int ct = 0; ct < TILES; ++ct) {
    int col = ct * 16 + nn;
#pragma unroll
    for (int r = 0; r < 8; ++r) {
      int grow = rowBase + wave * 16 + hi * 8 + r;
      if (grow < nrows && col < outCols) {
        float v = acc[ct][r] + bias[col];
        if (bn_g != nullptr) {
          v = (v - bn_m[col]) * rsqrtf(bn_v[col] + 1e-5f) * bn_g[col] + bn_b[col];
        }
        if (relu) v = fmaxf(v, 0.f);
        out[(long)grow * outCols + col] = v;
      }
    }
  }
}

// ---------------------------------------------------------------------------
// Residual VQ: wave32 per node, 4 dims per lane, codebook staged in LDS.
// loss += mean(res_final^2); ids[node, layer*3+g] = argmin code (as float).
// ---------------------------------------------------------------------------
__global__ void __launch_bounds__(256)
rvq_kernel(const float* __restrict__ h, const float* __restrict__ cb,
           float* __restrict__ loss, float* __restrict__ ids,
           int layer, int nrows) {
  __shared__ float scb[3 * 16 * 128];
  const int tid = threadIdx.x;
  for (int i = tid; i < 3 * 16 * 128; i += 256) scb[i] = cb[i];
  __syncthreads();

  const int wave = tid >> 5, lane = tid & 31;
  const int node = blockIdx.x * 8 + wave;
  if (node >= nrows) return;

  const float* hp = h + (long)node * 128 + lane * 4;
  float r0 = hp[0], r1 = hp[1], r2 = hp[2], r3 = hp[3];

  for (int g = 0; g < 3; ++g) {
    float bestd = 3.4e38f;
    int bestk = 0;
    for (int k = 0; k < 16; ++k) {
      const float* c = &scb[(g * 16 + k) * 128 + lane * 4];
      float dot = r0 * c[0] + r1 * c[1] + r2 * c[2] + r3 * c[3];
      float cn = c[0] * c[0] + c[1] * c[1] + c[2] * c[2] + c[3] * c[3];
#pragma unroll
      for (int off = 16; off >= 1; off >>= 1) {
        dot += __shfl_xor(dot, off, 32);
        cn += __shfl_xor(cn, off, 32);
      }
      float d2 = cn - 2.f * dot;  // |res|^2 constant over k: dropped
      if (d2 < bestd) { bestd = d2; bestk = k; }
    }
    const float* q = &scb[(g * 16 + bestk) * 128 + lane * 4];
    r0 -= q[0]; r1 -= q[1]; r2 -= q[2]; r3 -= q[3];
    if (lane == 0) ids[(long)node * 9 + layer * 3 + g] = (float)bestk;
  }
  float part = r0 * r0 + r1 * r1 + r2 * r2 + r3 * r3;
#pragma unroll
  for (int off = 16; off >= 1; off >>= 1) part += __shfl_xor(part, off, 32);
  if (lane == 0) atomicAdd(loss, part * (1.0f / ((float)nrows * 128.f)));
}

// ---------------------------------------------------------------------------
// Host-side orchestration
// ---------------------------------------------------------------------------
extern "C" void kernel_launch(void* const* d_in, const int* in_sizes, int n_in,
                              void* d_out, int out_size, void* d_ws, size_t ws_size,
                              hipStream_t stream) {
  const float* x    = (const float*)d_in[0];
  const int*   srcE = (const int*)d_in[1];
  const int*   dstE = (const int*)d_in[2];
  const float* W0a = (const float*)d_in[3];
  const float* W0r = (const float*)d_in[4];
  const float* b0  = (const float*)d_in[5];
  const float* W1a = (const float*)d_in[6];
  const float* W1r = (const float*)d_in[7];
  const float* b1  = (const float*)d_in[8];
  const float* W2a = (const float*)d_in[9];
  const float* W2r = (const float*)d_in[10];
  const float* b2  = (const float*)d_in[11];
  const float* bng = (const float*)d_in[12];
  const float* bnb = (const float*)d_in[13];
  const float* bnm = (const float*)d_in[14];
  const float* bnv = (const float*)d_in[15];
  const float* cb   = (const float*)d_in[16];
  const float* linW = (const float*)d_in[17];
  const float* linb = (const float*)d_in[18];

  const int N = in_sizes[0] / 100;
  const int E = in_sizes[1];

  char* ws = (char*)d_ws;
  size_t off = 0;
  float* hA  = (float*)(ws + off); off += (size_t)N * 128 * 4;
  float* hB  = (float*)(ws + off); off += (size_t)N * 128 * 4;
  float* agg = (float*)(ws + off); off += (size_t)N * 128 * 4;
  float* deg = (float*)(ws + off); off += (size_t)N * 4;
  off = (off + 255) & ~(size_t)255;
  __bf16* wA0 = (__bf16*)(ws + off); off += 4 * 8 * 512 * 2;
  __bf16* wR0 = (__bf16*)(ws + off); off += 4 * 8 * 512 * 2;
  __bf16* wA1 = (__bf16*)(ws + off); off += 4 * 8 * 512 * 2;
  __bf16* wR1 = (__bf16*)(ws + off); off += 4 * 8 * 512 * 2;
  __bf16* wA2 = (__bf16*)(ws + off); off += 4 * 8 * 512 * 2;
  __bf16* wR2 = (__bf16*)(ws + off); off += 4 * 8 * 512 * 2;
  __bf16* wLn = (__bf16*)(ws + off); off += 4 * 3 * 512 * 2;

  float* out  = (float*)d_out;
  float* loss = out + (size_t)N * 40;  // scalar after out[N,40]
  float* ids  = loss + 1;              // [N,9] as floats

  const long NH = (long)N * 128;
  dim3 blk(256);

  // Prep: pad x, pack weights into bf16 fragment order, zero loss accumulator
  pad_x_kernel<<<(unsigned)((NH + 255) / 256), blk, 0, stream>>>(x, hA, NH, N);
  pack_w_kernel<<<(4 * 8 * 512 + 255) / 256, blk, 0, stream>>>(W0a, 100, 128, wA0, 8);
  pack_w_kernel<<<(4 * 8 * 512 + 255) / 256, blk, 0, stream>>>(W0r, 100, 128, wR0, 8);
  pack_w_kernel<<<(4 * 8 * 512 + 255) / 256, blk, 0, stream>>>(W1a, 128, 128, wA1, 8);
  pack_w_kernel<<<(4 * 8 * 512 + 255) / 256, blk, 0, stream>>>(W1r, 128, 128, wR1, 8);
  pack_w_kernel<<<(4 * 8 * 512 + 255) / 256, blk, 0, stream>>>(W2a, 128, 128, wA2, 8);
  pack_w_kernel<<<(4 * 8 * 512 + 255) / 256, blk, 0, stream>>>(W2r, 128, 128, wR2, 8);
  pack_w_kernel<<<(4 * 3 * 512 + 255) / 256, blk, 0, stream>>>(linW, 128, 40, wLn, 3);
  zero_kernel<<<1, 1, 0, stream>>>(loss, 1);

  const float* inH[3]  = {hA, hB, hA};
  float*       outH[3] = {hB, hA, hB};
  const __bf16* Wa[3] = {wA0, wA1, wA2};
  const __bf16* Wr[3] = {wR0, wR1, wR2};
  const float* bias[3] = {b0, b1, b2};

  const long ET = (long)E * 32;
  for (int l = 0; l < 3; ++l) {
    zero_kernel<<<(unsigned)((NH + 255) / 256), blk, 0, stream>>>(agg, NH);
    zero_kernel<<<(unsigned)((N + 255) / 256), blk, 0, stream>>>(deg, N);
    deg_kernel<<<(unsigned)((E + 255) / 256), blk, 0, stream>>>(dstE, deg, E);
    scatter_kernel<<<(unsigned)((ET + 255) / 256), blk, 0, stream>>>(inH[l], srcE, dstE, agg, ET);
    div_kernel<<<(unsigned)((NH + 255) / 256), blk, 0, stream>>>(agg, deg, NH);

    const int relu = (l < 2) ? 1 : 0;
    const float* g = (l < 2) ? bng + l * 128 : nullptr;
    const float* bb = (l < 2) ? bnb + l * 128 : nullptr;
    const float* bm = (l < 2) ? bnm + l * 128 : nullptr;
    const float* bv = (l < 2) ? bnv + l * 128 : nullptr;
    sage_gemm<8, true><<<(unsigned)((N + 63) / 64), 128, 0, stream>>>(
        agg, inH[l], Wa[l], Wr[l], bias[l], g, bb, bm, bv, relu, N, outH[l], 128);

    rvq_kernel<<<(unsigned)((N + 7) / 8), blk, 0, stream>>>(
        outH[l], cb + (long)l * 3 * 16 * 128, loss, ids, l, N);
  }

  // Final linear: out[N,40] = h2 @ linW + linb (bf16 fragments padded to 48 cols)
  sage_gemm<3, false><<<(unsigned)((N + 63) / 64), 128, 0, stream>>>(
      hB, nullptr, wLn, nullptr, linb, nullptr, nullptr, nullptr, nullptr, 0, N, out, 40);
}
